// GlobalGGNN_74749610820333
// MI455X (gfx1250) — matrix operand order repaired and checked by reading, not compiled, not run on hardware
//
#include <hip/hip_runtime.h>
#include <hip/hip_bf16.h>
#include <math.h>
#include <stdint.h>

#define NN 20000
#define NE 320000
#define DD 256
#define LL 32
#define NG 64
#define OD 128
#define D3 768

typedef __attribute__((ext_vector_type(16))) __bf16 v16bf;
typedef __attribute__((ext_vector_type(8)))  float  v8f;
typedef __attribute__((ext_vector_type(4)))  int    v4i;

__device__ __forceinline__ unsigned short f32_to_bf16_bits(float f) {
  unsigned u = __float_as_uint(f);
  unsigned r = 0x7FFFu + ((u >> 16) & 1u);
  return (unsigned short)((u + r) >> 16);
}

// ---- CDNA5 async global->LDS staging (guarded; falls back to plain copy) ----
__device__ __forceinline__ void cp_b128(void* lds, const void* g) {
#if __has_builtin(__builtin_amdgcn_global_load_async_to_lds_b128)
  using gv4i_p = __attribute__((address_space(1))) v4i*;
  using lv4i_p = __attribute__((address_space(3))) v4i*;
  __builtin_amdgcn_global_load_async_to_lds_b128(
      (gv4i_p)(uintptr_t)g, (lv4i_p)(uintptr_t)lds, 0, 0);
#else
  *(uint4*)lds = *(const uint4*)g;
#endif
}

__device__ __forceinline__ void wait_async_all() {
#if __has_builtin(__builtin_amdgcn_s_wait_asynccnt)
  __builtin_amdgcn_s_wait_asynccnt(0);
#endif
}

// ---- fragment helpers -------------------------------------------------------
// A fragment from an LDS-resident bf16 panel stored pair-packed (uint = 2 halves
// along K). ldu = row stride in uints. ISA 16-bit A layout: lane L -> row L%16;
// lanes 0-15 K={0..7,16..23}, lanes 16-31 +8.
__device__ __forceinline__ v16bf load_frag_a_lds(const unsigned* __restrict__ As,
                                                 int ldu, int lane) {
  int row  = lane & 15;
  int koff = (lane >> 4) * 4;  // 8 halves = 4 uints
  union { v16bf v; unsigned u[8]; } r;
  const unsigned* base = As + row * ldu;
#pragma unroll
  for (int p = 0; p < 8; ++p) {
    int ku = ((p < 4) ? p : 8 + (p - 4)) + koff;  // uint index = k/2
    r.u[p] = base[ku];
  }
  return r.v;
}

// B fragment from pre-packed weights: per (ct, kb) tile, 32 lanes x 8 uints,
// contiguous per lane -> two coalesced 16B loads.
__device__ __forceinline__ v16bf load_frag_b_packed(const unsigned* __restrict__ P,
                                                    int kt, int ct, int kb, int lane) {
  const unsigned* src = P + ((size_t)(ct * kt + kb) * 32 + lane) * 8;
  union { v16bf v; uint4 q[2]; } r;
  r.q[0] = *(const uint4*)(src);
  r.q[1] = *(const uint4*)(src + 4);
  return r.v;
}

// C/D: lane L -> col L%16; VGPR r -> row r (lanes<16) or r+8 (lanes>=16).
__device__ __forceinline__ void store_tile(float* __restrict__ C, int ldc,
                                           v8f acc, int lane) {
  int n  = lane & 15;
  int mb = (lane >> 4) * 8;
#pragma unroll
  for (int r = 0; r < 8; ++r) C[(size_t)(mb + r) * ldc + n] = acc[r];
}

#define WMMA_BF16(a, b, c) \
  __builtin_amdgcn_wmma_f32_16x16x32_bf16(false, (a), false, (b), (short)0, (c), false, false)

// ---------------- weight packing (once per launch) ---------------------------
// Pack W[l][k][n] (B = W, K=256, N=256) into fragment layout:
// P[l][ct][kb][lane][p], k = kb*32 + (lane>>4)*8 + 2p + (p>=4 ? 8 : 0), n = ct*16 + (lane&15).
__global__ void k_pack_wm(const float* __restrict__ W, unsigned* __restrict__ P) {
  long long i = (long long)blockIdx.x * blockDim.x + threadIdx.x;
  if (i >= (long long)LL * 32768) return;
  int p    = (int)(i & 7);
  int lane = (int)((i >> 3) & 31);
  int kb   = (int)((i >> 8) & 7);
  int ct   = (int)((i >> 11) & 15);
  int l    = (int)(i >> 15);
  int k = kb * 32 + (lane >> 4) * 8 + 2 * p + (p >= 4 ? 8 : 0);
  int n = ct * 16 + (lane & 15);
  const float* Wl = W + (size_t)l * DD * DD;
  unsigned lo = f32_to_bf16_bits(Wl[(size_t)k * DD + n]);
  unsigned hi = f32_to_bf16_bits(Wl[(size_t)(k + 1) * DD + n]);
  P[i] = lo | (hi << 16);
}

// Pack w[j][k] ([768][256]) as B[k][j] = w[j][k] (K=256, N=768):
// P[ct][kb][lane][p], k = kb*32 + (lane>>4)*8 + 2p + (p>=4 ? 8 : 0), j = ct*16 + (lane&15).
// The (k,k+1) pair is contiguous in w's row.
__global__ void k_pack_wT(const float* __restrict__ w, unsigned* __restrict__ P) {
  int i = blockIdx.x * blockDim.x + threadIdx.x;
  if (i >= 48 * 8 * 32 * 8) return;
  int p    = i & 7;
  int lane = (i >> 3) & 31;
  int kb   = (i >> 8) & 7;
  int ct   = i >> 11;  // 0..47
  int k = kb * 32 + (lane >> 4) * 8 + 2 * p + (p >= 4 ? 8 : 0);
  int j = ct * 16 + (lane & 15);
  unsigned lo = f32_to_bf16_bits(w[(size_t)j * DD + k]);
  unsigned hi = f32_to_bf16_bits(w[(size_t)j * DD + k + 1]);
  P[i] = lo | (hi << 16);
}

// ---------------- misc elementwise kernels -----------------------------------
__global__ void k_cvt_bf16(const float* __restrict__ in, unsigned short* __restrict__ out,
                           long long n) {
  long long i = (long long)blockIdx.x * blockDim.x + threadIdx.x;
  if (i < n) out[i] = f32_to_bf16_bits(in[i]);
}

__global__ void k_init_h(const float* __restrict__ x, float* __restrict__ h,
                         unsigned short* __restrict__ hb, long long n) {
  long long i = (long long)blockIdx.x * blockDim.x + threadIdx.x;
  if (i < n) { float v = x[i]; h[i] = v; hb[i] = f32_to_bf16_bits(v); }
}

__global__ void k_zero(float* __restrict__ p, long long n) {
  long long i = (long long)blockIdx.x * blockDim.x + threadIdx.x;
  if (i < n) p[i] = 0.f;
}

// ---------------- per-layer kernels ------------------------------------------

// m = hb @ W_l. Block = 4 waves, one 16-row panel; wave computes a 16x64 strip
// (4 accumulators sharing each A fragment). K fully unrolled: 32 wmma/wave.
__global__ void k_gemm_m(const unsigned short* __restrict__ hb,
                         const unsigned* __restrict__ Wp,  // packed, this layer
                         float* __restrict__ m) {
  __shared__ unsigned As[2048];  // 16x256 bf16 A panel (pair-packed)
  int t = threadIdx.x;
  int lane = t & 31, wid = t >> 5;
  int rt = blockIdx.x;  // 0..1249
  const unsigned* gp = (const unsigned*)(hb + (size_t)rt * 16 * DD);
#pragma unroll
  for (int i = 0; i < 4; ++i) {
    int off = (i * 128 + t) * 4;
    cp_b128(&As[off], &gp[off]);
  }
  wait_async_all();
  __syncthreads();

  v8f acc0 = {0.f,0.f,0.f,0.f,0.f,0.f,0.f,0.f};
  v8f acc1 = acc0, acc2 = acc0, acc3 = acc0;
  int ct0 = wid * 4;
#pragma unroll
  for (int kb = 0; kb < 8; ++kb) {
    if (kb + 1 < 8)
      __builtin_prefetch(Wp + ((size_t)(ct0 * 8 + kb + 1) * 32) * 8, 0, 0);
    v16bf a = load_frag_a_lds(As + kb * 16, 128, lane);
    v16bf b0 = load_frag_b_packed(Wp, 8, ct0 + 0, kb, lane);
    acc0 = WMMA_BF16(a, b0, acc0);
    v16bf b1 = load_frag_b_packed(Wp, 8, ct0 + 1, kb, lane);
    acc1 = WMMA_BF16(a, b1, acc1);
    v16bf b2 = load_frag_b_packed(Wp, 8, ct0 + 2, kb, lane);
    acc2 = WMMA_BF16(a, b2, acc2);
    v16bf b3 = load_frag_b_packed(Wp, 8, ct0 + 3, kb, lane);
    acc3 = WMMA_BF16(a, b3, acc3);
  }
  float* Cb = m + (size_t)rt * 16 * DD;
  store_tile(Cb + (ct0 + 0) * 16, DD, acc0, lane);
  store_tile(Cb + (ct0 + 1) * 16, DD, acc1, lane);
  store_tile(Cb + (ct0 + 2) * 16, DD, acc2, lane);
  store_tile(Cb + (ct0 + 3) * 16, DD, acc3, lane);
}

// agg[dst] += m[src]
__global__ void k_scatter(const float* __restrict__ m, const int* __restrict__ src,
                          const int* __restrict__ dst, float* __restrict__ agg) {
  long long gid = (long long)blockIdx.x * blockDim.x + threadIdx.x;
  int e = (int)(gid >> 6);
  int q = (int)(gid & 63);
  if (e >= NE) return;
  int s = src[e], d = dst[e];
  float4 v = *(const float4*)(m + (size_t)s * DD + q * 4);
  float* p = agg + (size_t)d * DD + q * 4;
  unsafeAtomicAdd(p + 0, v.x);
  unsafeAtomicAdd(p + 1, v.y);
  unsafeAtomicAdd(p + 2, v.z);
  unsafeAtomicAdd(p + 3, v.w);
}

// Fused GRU. LDS: A panels for aggb & hb (staged once, async) + gi/gh staging.
__global__ void k_gru(const unsigned short* __restrict__ aggb,
                      unsigned short* __restrict__ hb,
                      float* __restrict__ h,
                      const unsigned* __restrict__ wihP,  // packed B, K=256 N=768
                      const unsigned* __restrict__ whhP,
                      const float* __restrict__ b_ih,
                      const float* __restrict__ b_hh) {
  extern __shared__ unsigned smemu[];
  unsigned* Agi_s = smemu;                    // 2048 uints = 16x256 bf16
  unsigned* Agh_s = smemu + 2048;
  float* gi = (float*)(smemu + 4096);         // 16 x 768
  float* gh = gi + 16 * D3;                   // 16 x 768
  int t = threadIdx.x, lane = t & 31, wid = t >> 5;
  int rt = blockIdx.x;  // 0..1249
  const unsigned* gpi = (const unsigned*)(aggb + (size_t)rt * 16 * DD);
  const unsigned* gph = (const unsigned*)(hb   + (size_t)rt * 16 * DD);
#pragma unroll
  for (int i = 0; i < 2; ++i) {
    int off = (i * 256 + t) * 4;
    cp_b128(&Agi_s[off], &gpi[off]);
    cp_b128(&Agh_s[off], &gph[off]);
  }
  wait_async_all();
  __syncthreads();

  for (int job = wid; job < 96; job += 8) {
    int isGh = (job >= 48);
    int ct   = isGh ? job - 48 : job;
    const unsigned* As = isGh ? Agh_s : Agi_s;
    const unsigned* P  = isGh ? whhP : wihP;
    v8f acc = {0.f,0.f,0.f,0.f,0.f,0.f,0.f,0.f};
#pragma unroll
    for (int kb = 0; kb < 8; ++kb) {
      v16bf a = load_frag_a_lds(As + kb * 16, 128, lane);
      v16bf b = load_frag_b_packed(P, 8, ct, kb, lane);
      acc = WMMA_BF16(a, b, acc);
    }
    float* dstp = (isGh ? gh : gi) + ct * 16;
    int n = lane & 15, mb = (lane >> 4) * 8;
#pragma unroll
    for (int r = 0; r < 8; ++r) dstp[(size_t)(mb + r) * D3 + n] = acc[r];
  }
  __syncthreads();

  for (int idx = threadIdx.x; idx < 16 * DD; idx += blockDim.x) {
    int row = idx >> 8, col = idx & 255;
    size_t node = (size_t)rt * 16 + row;
    float ir = gi[row * D3 + col]       + b_ih[col];
    float iz = gi[row * D3 + col + 256] + b_ih[col + 256];
    float in_= gi[row * D3 + col + 512] + b_ih[col + 512];
    float hr = gh[row * D3 + col]       + b_hh[col];
    float hz = gh[row * D3 + col + 256] + b_hh[col + 256];
    float hn = gh[row * D3 + col + 512] + b_hh[col + 512];
    float r = 1.f / (1.f + __expf(-(ir + hr)));
    float z = 1.f / (1.f + __expf(-(iz + hz)));
    float n2 = tanhf(in_ + r * hn);
    float ho = h[node * DD + col];
    float hv = (1.f - z) * n2 + z * ho;
    h[node * DD + col]  = hv;
    hb[node * DD + col] = f32_to_bf16_bits(hv);
  }
}

// ---------------- pooling / readout ------------------------------------------
__global__ void k_pool(const float* __restrict__ h, const int* __restrict__ batch,
                       float* __restrict__ sums, float* __restrict__ cnt) {
  int nidx = blockIdx.x;
  int c = threadIdx.x;
  int g = batch[nidx];
  unsafeAtomicAdd(&sums[(size_t)g * DD + c], h[(size_t)nidx * DD + c]);
  if (c == 0) unsafeAtomicAdd(&cnt[g], 1.f);
}

__global__ void k_final(const float* __restrict__ sums, const float* __restrict__ cnt,
                        const float* __restrict__ wf, const float* __restrict__ bf,
                        float* __restrict__ out) {
  __shared__ float pooled[DD];
  int g = blockIdx.x;
  float c = fmaxf(cnt[g], 1.f);
  for (int t = threadIdx.x; t < DD; t += blockDim.x)
    pooled[t] = sums[(size_t)g * DD + t] / c;
  __syncthreads();
  int j = threadIdx.x;  // 128 threads
  float acc = bf[j];
  for (int k = 0; k < DD; ++k) acc += pooled[k] * wf[(size_t)j * DD + k];
  out[(size_t)g * OD + j] = 1.f / (1.f + __expf(-acc));
}

// ---------------- host launcher ----------------------------------------------
extern "C" void kernel_launch(void* const* d_in, const int* in_sizes, int n_in,
                              void* d_out, int out_size, void* d_ws, size_t ws_size,
                              hipStream_t stream) {
  const float* x      = (const float*)d_in[0];
  const int*   ei     = (const int*)d_in[1];
  const int*   batch  = (const int*)d_in[2];
  const float* weight = (const float*)d_in[3];
  const float* w_ih   = (const float*)d_in[4];
  const float* w_hh   = (const float*)d_in[5];
  const float* b_ih   = (const float*)d_in[6];
  const float* b_hh   = (const float*)d_in[7];
  const float* w_fin  = (const float*)d_in[8];
  const float* b_fin  = (const float*)d_in[9];
  float* out = (float*)d_out;
  const int* src  = ei;
  const int* dstp = ei + NE;

  size_t off = 0;
  auto carve = [&](size_t bytes) -> void* {
    off = (off + 255) & ~(size_t)255;
    void* p = (char*)d_ws + off;
    off += bytes;
    return p;
  };
  float*          h    = (float*)carve((size_t)NN * DD * 4);
  unsigned short* hb   = (unsigned short*)carve((size_t)NN * DD * 2);
  float*          m    = (float*)carve((size_t)NN * DD * 4);
  float*          agg  = (float*)carve((size_t)NN * DD * 4);
  unsigned short* aggb = (unsigned short*)carve((size_t)NN * DD * 2);
  unsigned*       Wp   = (unsigned*)carve((size_t)LL * 32768 * 4);    // packed W
  unsigned*       wihP = (unsigned*)carve((size_t)48 * 8 * 32 * 8 * 4);
  unsigned*       whhP = (unsigned*)carve((size_t)48 * 8 * 32 * 8 * 4);
  float*          sums = (float*)carve((size_t)NG * DD * 4);
  float*          cnt  = (float*)carve((size_t)NG * 4);
  (void)ws_size; (void)in_sizes; (void)n_in; (void)out_size;

  long long nWm = (long long)LL * 32768;
  k_pack_wm<<<(unsigned)((nWm + 255) / 256), 256, 0, stream>>>(weight, Wp);
  int nWt = 48 * 8 * 32 * 8;
  k_pack_wT<<<(nWt + 255) / 256, 256, 0, stream>>>(w_ih, wihP);
  k_pack_wT<<<(nWt + 255) / 256, 256, 0, stream>>>(w_hh, whhP);
  long long nH = (long long)NN * DD;
  k_init_h<<<(unsigned)((nH + 255) / 256), 256, 0, stream>>>(x, h, hb, nH);

  const long long nSc = (long long)NE * 64;
  const unsigned gruLds = 4096u * 4u + 2u * 16u * D3 * 4u;  // 112 KB
  for (int l = 0; l < LL; ++l) {
    k_gemm_m<<<NN / 16, 128, 0, stream>>>(hb, Wp + (size_t)l * 32768, m);
    k_zero<<<(unsigned)((nH + 255) / 256), 256, 0, stream>>>(agg, nH);
    k_scatter<<<(unsigned)((nSc + 255) / 256), 256, 0, stream>>>(m, src, dstp, agg);
    k_cvt_bf16<<<(unsigned)((nH + 255) / 256), 256, 0, stream>>>(agg, aggb, nH);
    k_gru<<<NN / 16, 256, gruLds, stream>>>(aggb, hb, h, wihP, whhP, b_ih, b_hh);
  }

  k_zero<<<(NG * DD + 255) / 256, 256, 0, stream>>>(sums, NG * DD);
  k_zero<<<1, 64, 0, stream>>>(cnt, NG);
  k_pool<<<NN, DD, 0, stream>>>(h, batch, sums, cnt);
  k_final<<<NG, OD, 0, stream>>>(sums, cnt, w_fin, b_fin, out);
}
